// KohonenSOM_80247168958748
// MI455X (gfx1250) — matrix operand order repaired and checked by reading, not compiled, not run on hardware
//
#include <hip/hip_runtime.h>

// KohonenSOM distance matrix: out[b,n] = || x[b,:] - w[n,:] ||_2
// d2 = x_sq + w_sq - 2 * (x @ w^T); cross term via V_WMMA_F32_16X16X32_BF16.
//
// Fast path (needs ~18.2MB of d_ws):
//   prep_kernel:  x,w (f32) -> bf16 copies in ws + f32 row norms in ws
//   gemm kernel:  pure bf16 fragment loads + WMMA, f32 epilogue
// Fallback (ws too small): fused kernel converting on the fly with (__bf16) casts.

#define BATCH 32768
#define NN    2500
#define KD    256
#define BM    128   // block tile M
#define BN    64    // block tile N
#define TOTROWS (BATCH + NN)

typedef __attribute__((ext_vector_type(16))) __bf16 v16bf;
typedef __attribute__((ext_vector_type(8)))  __bf16 v8bf;
typedef __attribute__((ext_vector_type(8)))  float  v8f;

__device__ __forceinline__ v8f wmma_bf16(v16bf a, v16bf b, v8f c) {
  return __builtin_amdgcn_wmma_f32_16x16x32_bf16(
      /*neg_a=*/false, a, /*neg_b=*/false, b,
      /*c_mod=*/(short)0, c, /*reuse_a=*/false, /*reuse_b=*/false);
}

// ---------------- fast path: prep (convert + norms) ----------------
// One wave per 256-float row: each lane handles 8 consecutive elements
// (two b128 loads, one b128 bf16 store), then a wave32 reduction for ||.||^2.
__global__ __launch_bounds__(256)
void som_prep_kernel(const float* __restrict__ x, const float* __restrict__ w,
                     __bf16* __restrict__ xb, __bf16* __restrict__ wb,
                     float* __restrict__ xsq, float* __restrict__ wsq) {
  const int lane = threadIdx.x & 31;
  const int wave = threadIdx.x >> 5;
  const int row  = blockIdx.x * 8 + wave;
  if (row >= TOTROWS) return;

  const bool isx = row < BATCH;
  const int  r   = isx ? row : row - BATCH;
  const float*  srow = (isx ? x : w) + (size_t)r * KD;
  __bf16*       drow = (isx ? xb : wb) + (size_t)r * KD;

  float4 v0 = *(const float4*)(srow + lane * 8);
  float4 v1 = *(const float4*)(srow + lane * 8 + 4);

  float s = v0.x * v0.x + v0.y * v0.y + v0.z * v0.z + v0.w * v0.w
          + v1.x * v1.x + v1.y * v1.y + v1.z * v1.z + v1.w * v1.w;

  v8bf o;
  o[0] = (__bf16)v0.x; o[1] = (__bf16)v0.y; o[2] = (__bf16)v0.z; o[3] = (__bf16)v0.w;
  o[4] = (__bf16)v1.x; o[5] = (__bf16)v1.y; o[6] = (__bf16)v1.z; o[7] = (__bf16)v1.w;
  *(v8bf*)(drow + lane * 8) = o;

  #pragma unroll
  for (int m = 16; m >= 1; m >>= 1) s += __shfl_xor(s, m, 32);
  if (lane == 0) { if (isx) xsq[r] = s; else wsq[r] = s; }
}

// A fragment (16x32 bf16, MxK): lane holds row m; element e -> K = kb + e,
// kb = 16*(lane>=16). One contiguous 32B run -> 2 x b128, then register concat.
__device__ __forceinline__ v16bf ldA(const __bf16* __restrict__ row, int kb) {
  v8bf lo = *(const v8bf*)(row + kb);
  v8bf hi = *(const v8bf*)(row + kb + 8);
  return __builtin_shufflevector(lo, hi, 0,1,2,3,4,5,6,7,8,9,10,11,12,13,14,15);
}
// B fragment (32x16 bf16, KxN): lane holds col n (= a W row); element e -> K = ko + e,
// ko = 16*(lane>=16). One contiguous 32B run -> 2 x b128.
__device__ __forceinline__ v16bf ldB(const __bf16* __restrict__ row, int ko) {
  v8bf lo = *(const v8bf*)(row + ko);
  v8bf hi = *(const v8bf*)(row + ko + 8);
  return __builtin_shufflevector(lo, hi, 0,1,2,3,4,5,6,7,8,9,10,11,12,13,14,15);
}

// ---------------- fast path: bf16 GEMM + distance epilogue ----------------
__global__ __launch_bounds__(256)
void som_dist_bf16_kernel(const __bf16* __restrict__ xb, const __bf16* __restrict__ wb,
                          const float* __restrict__ xsq, const float* __restrict__ wsq,
                          float* __restrict__ out) {
  const int tid   = threadIdx.x;
  const int n_blk = blockIdx.x * BN;
  const int m_blk = blockIdx.y * BM;

  const int lane = tid & 31;
  const int wave = tid >> 5;
  const int wm   = wave >> 1;     // 0..3 -> 32-row slab
  const int wn   = wave & 1;      // 0..1 -> 32-col slab
  const int lrow = lane & 15;
  const int hi   = lane >> 4;     // half-wave

  const int m_base = m_blk + wm * 32;
  const int n_base = n_blk + wn * 32;

  const __bf16* arow0 = xb + (size_t)(m_base +      lrow) * KD;
  const __bf16* arow1 = xb + (size_t)(m_base + 16 + lrow) * KD;

  const int n0  = n_base +      lrow;
  const int n1  = n_base + 16 + lrow;
  const int n0c = (n0 < NN) ? n0 : (NN - 1);   // clamp: loads stay in-bounds,
  const int n1c = (n1 < NN) ? n1 : (NN - 1);   // EXEC stays all-ones for WMMA
  const __bf16* brow0 = wb + (size_t)n0c * KD;
  const __bf16* brow1 = wb + (size_t)n1c * KD;

  const int akoff = hi * 16;
  const int bkoff = hi * 16;

  v8f acc00 = {}, acc01 = {}, acc10 = {}, acc11 = {};

  #pragma unroll
  for (int k0 = 0; k0 < KD; k0 += 32) {
    v16bf a0 = ldA(arow0, k0 + akoff);
    v16bf a1 = ldA(arow1, k0 + akoff);
    v16bf b0 = ldB(brow0, k0 + bkoff);
    v16bf b1 = ldB(brow1, k0 + bkoff);
    acc00 = wmma_bf16(a0, b0, acc00);
    acc01 = wmma_bf16(a0, b1, acc01);
    acc10 = wmma_bf16(a1, b0, acc10);
    acc11 = wmma_bf16(a1, b1, acc11);
  }

  // C/D layout: VGPR r, lane L -> m_sub = r + 8*hi, n_sub = L%16.
  const float wsq0 = wsq[n0c];
  const float wsq1 = wsq[n1c];

  #pragma unroll
  for (int msub = 0; msub < 2; ++msub) {
    const v8f an0 = msub ? acc10 : acc00;
    const v8f an1 = msub ? acc11 : acc01;
    const int mabs = m_blk + wm * 32 + msub * 16 + hi * 8;
    float4 q0 = *(const float4*)(xsq + mabs);
    float4 q1 = *(const float4*)(xsq + mabs + 4);
    const float xs[8] = {q0.x, q0.y, q0.z, q0.w, q1.x, q1.y, q1.z, q1.w};
    #pragma unroll
    for (int r = 0; r < 8; ++r) {
      const size_t mrow = (size_t)(mabs + r) * NN;
      const float d0 = __builtin_sqrtf(fmaxf(xs[r] + wsq0 - 2.0f * an0[r], 0.0f));
      const float d1 = __builtin_sqrtf(fmaxf(xs[r] + wsq1 - 2.0f * an1[r], 0.0f));
      if (n0 < NN) out[mrow + n0] = d0;
      if (n1 < NN) out[mrow + n1] = d1;
    }
  }
}

// ---------------- fallback: fused, converts on the fly ----------------
__device__ __forceinline__ v16bf cvt16(float4 a, float4 b, float4 c, float4 d) {
  v16bf u;
  u[0]  = (__bf16)a.x; u[1]  = (__bf16)a.y; u[2]  = (__bf16)a.z; u[3]  = (__bf16)a.w;
  u[4]  = (__bf16)b.x; u[5]  = (__bf16)b.y; u[6]  = (__bf16)b.z; u[7]  = (__bf16)b.w;
  u[8]  = (__bf16)c.x; u[9]  = (__bf16)c.y; u[10] = (__bf16)c.z; u[11] = (__bf16)c.w;
  u[12] = (__bf16)d.x; u[13] = (__bf16)d.y; u[14] = (__bf16)d.z; u[15] = (__bf16)d.w;
  return u;
}
__device__ __forceinline__ v16bf ldA_f32(const float* __restrict__ row, int kb) {
  const float4* p0 = (const float4*)(row + kb);
  const float4* p1 = (const float4*)(row + kb + 8);
  return cvt16(p0[0], p0[1], p1[0], p1[1]);
}
__device__ __forceinline__ v16bf ldB_f32(const float* __restrict__ row, int ko) {
  const float4* p = (const float4*)(row + ko);
  return cvt16(p[0], p[1], p[2], p[3]);
}

__global__ __launch_bounds__(256)
void som_dist_fused_kernel(const float* __restrict__ x, const float* __restrict__ w,
                           float* __restrict__ out) {
  __shared__ float s_xsq[BM];
  __shared__ float s_wsq[BN];

  const int tid   = threadIdx.x;
  const int n_blk = blockIdx.x * BN;
  const int m_blk = blockIdx.y * BM;

  if (tid < BM) {
    const float* r = x + (size_t)(m_blk + tid) * KD;
    float s = 0.f;
    #pragma unroll 4
    for (int k = 0; k < KD; k += 4) {
      float4 v = *(const float4*)(r + k);
      s += v.x * v.x + v.y * v.y + v.z * v.z + v.w * v.w;
    }
    s_xsq[tid] = s;
  } else if (tid < BM + BN) {
    const int j = tid - BM;
    int n = n_blk + j; if (n > NN - 1) n = NN - 1;
    const float* r = w + (size_t)n * KD;
    float s = 0.f;
    #pragma unroll 4
    for (int k = 0; k < KD; k += 4) {
      float4 v = *(const float4*)(r + k);
      s += v.x * v.x + v.y * v.y + v.z * v.z + v.w * v.w;
    }
    s_wsq[j] = s;
  }
  __syncthreads();

  const int lane = tid & 31;
  const int wave = tid >> 5;
  const int wm = wave >> 1, wn = wave & 1;
  const int lrow = lane & 15, hi = lane >> 4;

  const int m_base = m_blk + wm * 32;
  const int n_base = n_blk + wn * 32;
  const float* arow0 = x + (size_t)(m_base +      lrow) * KD;
  const float* arow1 = x + (size_t)(m_base + 16 + lrow) * KD;
  const int n0 = n_base + lrow, n1 = n_base + 16 + lrow;
  const int n0c = (n0 < NN) ? n0 : (NN - 1);
  const int n1c = (n1 < NN) ? n1 : (NN - 1);
  const float* brow0 = w + (size_t)n0c * KD;
  const float* brow1 = w + (size_t)n1c * KD;
  const int akoff = hi * 16, bkoff = hi * 16;

  v8f acc00 = {}, acc01 = {}, acc10 = {}, acc11 = {};
  #pragma unroll
  for (int k0 = 0; k0 < KD; k0 += 32) {
    v16bf a0 = ldA_f32(arow0, k0 + akoff);
    v16bf a1 = ldA_f32(arow1, k0 + akoff);
    v16bf b0 = ldB_f32(brow0, k0 + bkoff);
    v16bf b1 = ldB_f32(brow1, k0 + bkoff);
    acc00 = wmma_bf16(a0, b0, acc00);
    acc01 = wmma_bf16(a0, b1, acc01);
    acc10 = wmma_bf16(a1, b0, acc10);
    acc11 = wmma_bf16(a1, b1, acc11);
  }

  const float wsq0 = s_wsq[wn * 32 + lrow];
  const float wsq1 = s_wsq[wn * 32 + 16 + lrow];
  #pragma unroll
  for (int msub = 0; msub < 2; ++msub) {
    const v8f an0 = msub ? acc10 : acc00;
    const v8f an1 = msub ? acc11 : acc01;
    const int mrel = wm * 32 + msub * 16 + hi * 8;
    #pragma unroll
    for (int r = 0; r < 8; ++r) {
      const float  xsv  = s_xsq[mrel + r];
      const size_t mrow = (size_t)(m_blk + mrel + r) * NN;
      const float d0 = __builtin_sqrtf(fmaxf(xsv + wsq0 - 2.0f * an0[r], 0.0f));
      const float d1 = __builtin_sqrtf(fmaxf(xsv + wsq1 - 2.0f * an1[r], 0.0f));
      if (n0 < NN) out[mrow + n0] = d0;
      if (n1 < NN) out[mrow + n1] = d1;
    }
  }
}

extern "C" void kernel_launch(void* const* d_in, const int* in_sizes, int n_in,
                              void* d_out, int out_size, void* d_ws, size_t ws_size,
                              hipStream_t stream) {
  (void)in_sizes; (void)n_in; (void)out_size;
  const float* x   = (const float*)d_in[0];   // [BATCH, KD]
  const float* w   = (const float*)d_in[1];   // [NN, KD]
  float*       out = (float*)d_out;           // [BATCH, NN]

  // ws layout (bytes): x_bf | w_bf | x_sq | w_sq  (all 256B-aligned)
  const size_t off_xb  = 0;
  const size_t off_wb  = off_xb + (size_t)BATCH * KD * 2;   // 16,777,216
  const size_t off_xsq = off_wb + (size_t)NN * KD * 2;      // 18,057,216
  const size_t off_wsq = off_xsq + (size_t)BATCH * 4;       // 18,188,288
  const size_t ws_need = off_wsq + (size_t)NN * 4;          // 18,198,288

  dim3 grid((NN + BN - 1) / BN, BATCH / BM);  // (40, 256)

  if (ws_size >= ws_need) {
    char* ws = (char*)d_ws;
    __bf16* xb  = (__bf16*)(ws + off_xb);
    __bf16* wb  = (__bf16*)(ws + off_wb);
    float*  xsq = (float*)(ws + off_xsq);
    float*  wsq = (float*)(ws + off_wsq);
    som_prep_kernel<<<(TOTROWS + 7) / 8, dim3(256), 0, stream>>>(x, w, xb, wb, xsq, wsq);
    som_dist_bf16_kernel<<<grid, dim3(256), 0, stream>>>(xb, wb, xsq, wsq, out);
  } else {
    som_dist_fused_kernel<<<grid, dim3(256), 0, stream>>>(x, w, out);
  }
}
